// Elmo_33921651704207
// MI455X (gfx1250) — compile-verified, matrix operand order checked by reading, not run on hardware
//
#include <hip/hip_runtime.h>

typedef __attribute__((ext_vector_type(16))) __bf16 v16bf;
typedef __attribute__((ext_vector_type(8)))  float  v8f;
typedef __attribute__((ext_vector_type(4)))  float  v4f;

#define LAYERS 2
#define BATCH  4
#define SEQ    64
#define DIN    512
#define HID    512
#define CELL   4096
#define GATESZ (4*CELL)   /* 16384 */
#define CLIPV  3.0f

#define NBLK   128
#define NBD    64
#define NTHR   256
#define WPD    (NBD*8)    /* waves per direction = 512 */

/* ---- workspace layout (bytes) ---- */
#define OFF_WSBF   0ull
#define SZ_WSBF    (4ull*GATESZ*HID*2ull)           /* 64 MB packed bf16 Ws */
#define OFF_WPBF   (OFF_WSBF + SZ_WSBF)
#define SZ_WPBF    (4ull*HID*CELL*2ull)             /* 16 MB packed bf16 Wp */
#define OFF_PI     (OFF_WPBF + SZ_WPBF)
#define SZ_PI      (2ull*SEQ*GATESZ*BATCH*4ull)     /* 32 MB, [dir][t][g][b] */
#define OFF_GATES  (OFF_PI + SZ_PI)
#define SZ_GATES   (2ull*GATESZ*BATCH*4ull)         /* [dir][g][b] */
#define OFF_CST    (OFF_GATES + SZ_GATES)
#define SZ_CST     (2ull*BATCH*CELL*4ull)
#define OFF_HST    (OFF_CST + SZ_CST)
#define SZ_HST     (2ull*BATCH*HID*4ull)
#define OFF_HACC   (OFF_HST + SZ_HST)
#define SZ_HACC    SZ_HST
#define OFF_RBUF   (OFF_HACC + SZ_HACC)
#define SZ_RBUF    (2ull*BATCH*SEQ*HID*4ull)
#define OFF_FWD    (OFF_RBUF + SZ_RBUF)
#define SZ_FWD     ((unsigned long long)BATCH*SEQ*HID*4ull)
#define OFF_BWD    (OFF_FWD + SZ_FWD)
#define OFF_XBF    (OFF_BWD + SZ_FWD)
#define SZ_XBF     (2ull*BATCH*SEQ*DIN*2ull)        /* packed bf16 x, per dir */
#define OFF_HBF    (OFF_XBF + SZ_XBF)
#define SZ_HBF     (2ull*16*32*16*2ull)             /* 16 KB/dir packed h frags */
#define OFF_PBF    (OFF_HBF + SZ_HBF)
#define SZ_PBF     (2ull*128*32*16*2ull)            /* 128 KB/dir packed p frags */
#define OFF_SYNC   (OFF_PBF + SZ_PBF)

__device__ __forceinline__ v8f wmma_bf16(v16bf a, v16bf b, v8f c) {
    return __builtin_amdgcn_wmma_f32_16x16x32_bf16(false, a, false, b,
                                                   (short)0, c, false, false);
}

__device__ __forceinline__ float sigf(float x) { return 1.0f / (1.0f + __expf(-x)); }
__device__ __forceinline__ float clip3(float x) { return fminf(fmaxf(x, -CLIPV), CLIPV); }

/* A/B fragment K mapping: frag elem e (lane-half kh) covers
   k = (e<8) ? kh*8+e : 16+kh*8+(e-8), within a 32-wide K tile.            */
__device__ __forceinline__ void frag_pos(int k_in_tile, int b_or_m, int* lane, int* e) {
    int kh, ee;
    if (k_in_tile < 16) { kh = k_in_tile >> 3; ee = k_in_tile & 7; }
    else { int u = k_in_tile - 16; kh = u >> 3; ee = 8 + (u & 7); }
    *lane = kh * 16 + b_or_m;
    *e = ee;
}

/* Build a 16x32 bf16 fragment from an fp32 row-major matrix (one-shot paths). */
__device__ __forceinline__ v16bf load_frag_f32(const float* base, int ld, int row,
                                               int k0, int kh) {
    v16bf r;
    const float* p = base + (size_t)row * ld + k0 + kh * 8;
#pragma unroll
    for (int j = 0; j < 8; ++j) { r[j] = (__bf16)p[j]; r[j + 8] = (__bf16)p[j + 16]; }
    return r;
}

/* Device-scope grid barrier (all NBLK blocks co-resident; persistent kernel). */
__device__ __forceinline__ void grid_sync(unsigned* cnt, unsigned* gen, unsigned nb) {
    __syncthreads();
    if (threadIdx.x == 0) {
        __threadfence();
        unsigned g = __atomic_load_n(gen, __ATOMIC_ACQUIRE);
        if (atomicAdd(cnt, 1u) == nb - 1u) {
            __atomic_store_n(cnt, 0u, __ATOMIC_RELAXED);
            __atomic_fetch_add(gen, 1u, __ATOMIC_RELEASE);
        } else {
            while (__atomic_load_n(gen, __ATOMIC_ACQUIRE) == g)
                __builtin_amdgcn_s_sleep(2);
        }
    }
    __syncthreads();
}

/* fp32 (G x K) row-major -> packed bf16 WMMA B-fragments.
   Tile (nt,kt) = 32 lanes x 16 bf16 contiguous (1 KB).                      */
__global__ void pack_weights(const float* __restrict__ W, __bf16* __restrict__ out,
                             int G, int K, int nmat) {
    const int KT = K / 32;
    const size_t per = (size_t)G * K;
    const size_t total = (size_t)nmat * per;
    for (size_t i = (size_t)blockIdx.x * blockDim.x + threadIdx.x; i < total;
         i += (size_t)gridDim.x * blockDim.x) {
        int mat = (int)(i / per);
        size_t r = i % per;
        int elem = (int)(r & 15);
        int lane = (int)((r >> 4) & 31);
        int tile = (int)(r >> 9);
        int kt = tile % KT, nt = tile / KT;
        int n = nt * 16 + (lane & 15);
        int kh = lane >> 4;
        int k = kt * 32 + ((elem < 8) ? (kh * 8 + elem) : (16 + kh * 8 + (elem - 8)));
        out[i] = (__bf16)W[(size_t)mat * per + (size_t)n * K + k];
    }
}

/* fp32 activations (256 x 512 per dir) -> packed bf16 A-fragments. */
__global__ void pack_x(const float* __restrict__ fwd, const float* __restrict__ bwd,
                       __bf16* __restrict__ out) {
    int i = blockIdx.x * blockDim.x + threadIdx.x;
    if (i >= 2 * BATCH * SEQ * DIN) return;
    int dir = i >> 17;
    int r = i & (BATCH * SEQ * DIN - 1);
    int e = r & 15, lane = (r >> 4) & 31, tile = r >> 9;
    int kt = tile & 15, mt = tile >> 4;
    int row = mt * 16 + (lane & 15);
    int kh = lane >> 4;
    int k = kt * 32 + ((e < 8) ? (kh * 8 + e) : (16 + kh * 8 + (e - 8)));
    const float* x = dir ? bwd : fwd;
    out[i] = (__bf16)x[(size_t)row * DIN + k];
}

__global__ void init_kernel(const float* __restrict__ inp, float* fwd, float* bwd,
                            unsigned* sync) {
    int i = blockIdx.x * blockDim.x + threadIdx.x;
    if (i < BATCH * SEQ * DIN) { float v = inp[i]; fwd[i] = v; bwd[i] = v; }
    if (i < 8) sync[i] = 0u;
}

/* pi = x @ Wi.T. One wave: 4 M-tiles x 1 N-tile; output [dir][t][g][b].    */
__global__ __launch_bounds__(NTHR) void gemm_pi_kernel(const __bf16* __restrict__ xbf,
                                                       const float* __restrict__ Wi,
                                                       float* __restrict__ pi, int layer) {
    int w = blockIdx.x * (NTHR >> 5) + (threadIdx.x >> 5);
    int lane = threadIdx.x & 31, nl = lane & 15, kh = lane >> 4;
    int dir = w >> 12;
    int rem = w & 4095;
    int mg = rem >> 10;
    int nt = rem & 1023;
    const __bf16* xb = xbf + (size_t)dir * BATCH * SEQ * DIN;
    const float* Wm = Wi + (size_t)(layer * 2 + dir) * GATESZ * DIN;
    v8f acc[4];
#pragma unroll
    for (int j = 0; j < 4; ++j)
#pragma unroll
        for (int r = 0; r < 8; ++r) acc[j][r] = 0.0f;
#pragma unroll
    for (int kt = 0; kt < DIN / 32; ++kt) {
        v16bf b = load_frag_f32(Wm, DIN, nt * 16 + nl, kt * 32, kh);
#pragma unroll
        for (int j = 0; j < 4; ++j) {
            int mt = mg * 4 + j;
            v16bf a = *(const v16bf*)(xb + (((size_t)mt * 16 + kt) * 32 + lane) * 16);
            acc[j] = wmma_bf16(a, b, acc[j]);
        }
    }
    float* po = pi + (size_t)dir * SEQ * GATESZ * BATCH;
    int col = nt * 16 + nl;
#pragma unroll
    for (int j = 0; j < 4; ++j) {
#pragma unroll
        for (int r = 0; r < 8; ++r) {
            int bt = (mg * 4 + j) * 16 + r + 8 * kh;
            int b = bt >> 6, t = bt & 63;
            po[((size_t)t * GATESZ + col) * 4 + b] = acc[j][r];
        }
    }
}

/* Persistent 64-step LSTM scan; both directions share one grid. */
__global__ __launch_bounds__(NTHR) void scan_kernel(
    const float* __restrict__ pi, const int* __restrict__ mask,
    const float* __restrict__ bs, const __bf16* __restrict__ wsbf,
    const __bf16* __restrict__ wpbf, float* gates, float* cst,
    float* hst, float* hacc, float* rbuf, __bf16* hbf, __bf16* pbf,
    unsigned* sync, int layer) {
    __shared__ __attribute__((aligned(16))) __bf16 hlds[16 * 32 * 16]; /* 16 KB @ LDS 0 */

    int tid = threadIdx.x, blk = blockIdx.x;
    int dir = (blk >= NBD) ? 1 : 0;
    int blkd = blk - dir * NBD;
    int wave_d = blkd * 8 + (tid >> 5);
    int lane = tid & 31, nl = lane & 15, kh = lane >> 4;
    int idxd = blkd * NTHR + tid;

    float* G  = gates + (size_t)dir * GATESZ * BATCH;
    float* Cs = cst   + (size_t)dir * BATCH * CELL;
    float* Hs = hst   + (size_t)dir * BATCH * HID;
    float* Ha = hacc  + (size_t)dir * BATCH * HID;
    float* R  = rbuf  + (size_t)dir * BATCH * SEQ * HID;
    __bf16* Hb = hbf  + (size_t)dir * 16 * 32 * 16;
    __bf16* Pb = pbf  + (size_t)dir * 128 * 32 * 16;
    const float*  PI  = pi   + (size_t)dir * SEQ * GATESZ * BATCH;
    const __bf16* WsM = wsbf + (size_t)(layer * 2 + dir) * GATESZ * HID;
    const __bf16* WpM = wpbf + (size_t)(layer * 2 + dir) * HID * CELL;
    const float*  bsv = bs   + (size_t)(layer * 2 + dir) * GATESZ;
    unsigned* cnt = &sync[0]; unsigned* gen = &sync[1];

    /* reset per-layer recurrent state + packed fragment buffers */
    Cs[idxd] = 0.0f;
    if (idxd < BATCH * HID) { Hs[idxd] = 0.0f; }
    ((unsigned*)Pb)[idxd] = 0u;
    ((unsigned*)Pb)[idxd + 16384] = 0u;
    if (idxd < 4096) ((unsigned*)Hb)[idxd] = 0u;
    grid_sync(cnt, gen, NBLK);

    for (int t = 0; t < SEQ; ++t) {
        int tt = dir ? (SEQ - 1 - t) : t;

        /* stage packed h fragments into LDS via async memory->LDS copies
           (GLOBAL_LOAD_ASYNC_TO_LDS_B128, ASYNCcnt). hlds sits at LDS offset
           0; pass hlds's generic pointer into the asm so the shared array
           escapes and the memory clobber is known to write it (keeps the
           ds_load_b128 consumers alive).                                     */
        {
            void* lesc = (void*)hlds;   /* provenance escape for hlds */
#pragma unroll
            for (int i = 0; i < 4; ++i) {
                unsigned laddr = (unsigned)(tid + i * NTHR) * 16u;
                unsigned long long gaddr =
                    (unsigned long long)(const char*)Hb + (size_t)laddr;
                asm volatile("global_load_async_to_lds_b128 %0, %1, off"
                             :: "v"(laddr), "v"(gaddr), "v"(lesc) : "memory");
            }
            asm volatile("s_wait_asynccnt 0x0" ::: "memory");
        }
        __syncthreads();

        /* phase 1: gates = pi_t + bias + h @ Ws.T
           one wave owns two N-tiles (nt0, nt0+512) sharing the A fragment   */
        {
            int nt0 = wave_d, nt1 = wave_d + WPD;
            int col0 = nt0 * 16 + nl, col1 = nt1 * 16 + nl;
            v8f acc0, acc1;
#pragma unroll
            for (int r = 0; r < 8; ++r) { acc0[r] = 0.0f; acc1[r] = 0.0f; }
            if (kh == 0) {
                v4f c0 = *(const v4f*)(PI + ((size_t)tt * GATESZ + col0) * 4);
                v4f c1 = *(const v4f*)(PI + ((size_t)tt * GATESZ + col1) * 4);
                float b0 = bsv[col0], b1 = bsv[col1];
#pragma unroll
                for (int r = 0; r < 4; ++r) { acc0[r] = c0[r] + b0; acc1[r] = c1[r] + b1; }
            }
#pragma unroll
            for (int kt = 0; kt < HID / 32; ++kt) {
                v16bf a = *(const v16bf*)(hlds + ((size_t)kt * 32 + lane) * 16);
                const __bf16* bp0 =
                    WsM + (((size_t)nt0 * (HID / 32) + kt) * 32 + lane) * 16;
                const __bf16* bp1 =
                    WsM + (((size_t)nt1 * (HID / 32) + kt) * 32 + lane) * 16;
                __builtin_prefetch(bp0 + 512, 0, 1);
                __builtin_prefetch(bp1 + 512, 0, 1);
                v16bf b0 = *(const v16bf*)bp0;
                v16bf b1 = *(const v16bf*)bp1;
                acc0 = wmma_bf16(a, b0, acc0);
                acc1 = wmma_bf16(a, b1, acc1);
            }
            if (kh == 0) {
                v4f o0, o1;
#pragma unroll
                for (int r = 0; r < 4; ++r) { o0[r] = acc0[r]; o1[r] = acc1[r]; }
                *(v4f*)(G + (size_t)col0 * 4) = o0;
                *(v4f*)(G + (size_t)col1 * 4) = o1;
            }
        }
        grid_sync(cnt, gen, NBLK);

        /* phase 2: cell update; write p directly as packed bf16 A-fragments */
        {
            int b = idxd >> 12, n = idxd & (CELL - 1);
            float gi = G[(size_t)n * 4 + b];
            float gf = G[((size_t)CELL + n) * 4 + b];
            float gg = G[((size_t)2 * CELL + n) * 4 + b];
            float go = G[((size_t)3 * CELL + n) * 4 + b];
            float cold = Cs[idxd];
            float cn = clip3(sigf(gi) * tanhf(gg) + sigf(gf) * cold);
            bool valid = mask[b * SEQ + tt] > 0;
            Cs[idxd] = valid ? cn : cold;
            float pv = sigf(go) * tanhf(cn);
            int l2, e2;
            frag_pos(n & 31, b, &l2, &e2);
            Pb[(((size_t)(n >> 5)) * 32 + l2) * 16 + e2] = (__bf16)pv;
            if (idxd < BATCH * HID) Ha[idxd] = 0.0f;
        }
        grid_sync(cnt, gen, NBLK);

        /* phase 3: h_acc = p @ Wp.T  (32 N-tiles x 16-way K split = 512 waves) */
        {
            int ntile = wave_d >> 4, ks = wave_d & 15;
            v8f acc;
#pragma unroll
            for (int r = 0; r < 8; ++r) acc[r] = 0.0f;
#pragma unroll
            for (int kt = ks * 8; kt < ks * 8 + 8; ++kt) {
                v16bf a = *(const v16bf*)(Pb + ((size_t)kt * 32 + lane) * 16);
                const __bf16* bp =
                    WpM + (((size_t)ntile * (CELL / 32) + kt) * 32 + lane) * 16;
                v16bf b = *(const v16bf*)bp;
                acc = wmma_bf16(a, b, acc);
            }
#pragma unroll
            for (int r = 0; r < 8; ++r) {
                int m = r + 8 * kh;
                if (m < BATCH)
                    atomicAdd(&Ha[(size_t)m * HID + ntile * 16 + nl], acc[r]);
            }
        }
        grid_sync(cnt, gen, NBLK);

        /* phase 4: clip/mask h, emit output row, write packed h fragments */
        if (idxd < BATCH * HID) {
            int b = idxd >> 9, n = idxd & (HID - 1);
            float hn = clip3(Ha[idxd]);
            bool valid = mask[b * SEQ + tt] > 0;
            float hv = valid ? hn : Hs[idxd];
            Hs[idxd] = hv;
            R[((size_t)b * SEQ + tt) * HID + n] = valid ? hn : 0.0f;
            int l2, e2;
            frag_pos(n & 31, b, &l2, &e2);
            Hb[(((size_t)(n >> 5)) * 32 + l2) * 16 + e2] = (__bf16)hv;
        }
        grid_sync(cnt, gen, NBLK);
    }
}

/* skip connection + write (L,B,T,2H) output and next-layer inputs */
__global__ void finalize_kernel(const float* __restrict__ rbuf, float* fwd, float* bwd,
                                float* __restrict__ out, int layer) {
    int i = blockIdx.x * blockDim.x + threadIdx.x;
    if (i >= 2 * BATCH * SEQ * HID) return;
    int dir = i >> 17;
    int r = i & (BATCH * SEQ * HID - 1);
    float v = rbuf[(size_t)dir * BATCH * SEQ * HID + r];
    float* io = dir ? bwd : fwd;
    if (layer > 0) v += io[r];
    int b = r >> 15, rem = r & 32767, t = rem >> 9, n = rem & 511;
    out[(((size_t)layer * BATCH + b) * SEQ + t) * (2 * HID) + dir * HID + n] = v;
    io[r] = v;
}

extern "C" void kernel_launch(void* const* d_in, const int* in_sizes, int n_in,
                              void* d_out, int out_size, void* d_ws, size_t ws_size,
                              hipStream_t stream) {
    const float* inputs = (const float*)d_in[0];
    const int*   mask   = (const int*)d_in[1];
    const float* Wi     = (const float*)d_in[2];
    const float* Ws     = (const float*)d_in[3];
    const float* bs     = (const float*)d_in[4];
    const float* Wp     = (const float*)d_in[5];
    float* out = (float*)d_out;
    char* ws = (char*)d_ws;

    __bf16* wsbf = (__bf16*)(ws + OFF_WSBF);
    __bf16* wpbf = (__bf16*)(ws + OFF_WPBF);
    float* pi    = (float*)(ws + OFF_PI);
    float* gates = (float*)(ws + OFF_GATES);
    float* cst   = (float*)(ws + OFF_CST);
    float* hst   = (float*)(ws + OFF_HST);
    float* hacc  = (float*)(ws + OFF_HACC);
    float* rbuf  = (float*)(ws + OFF_RBUF);
    float* fwd   = (float*)(ws + OFF_FWD);
    float* bwd   = (float*)(ws + OFF_BWD);
    __bf16* xbf  = (__bf16*)(ws + OFF_XBF);
    __bf16* hbf  = (__bf16*)(ws + OFF_HBF);
    __bf16* pbf  = (__bf16*)(ws + OFF_PBF);
    unsigned* sync = (unsigned*)(ws + OFF_SYNC);

    init_kernel<<<512, 256, 0, stream>>>(inputs, fwd, bwd, sync);
    pack_weights<<<4096, 256, 0, stream>>>(Ws, wsbf, GATESZ, HID, 4);
    pack_weights<<<2048, 256, 0, stream>>>(Wp, wpbf, HID, CELL, 4);

    for (int l = 0; l < LAYERS; ++l) {
        pack_x<<<1024, 256, 0, stream>>>(fwd, bwd, xbf);
        gemm_pi_kernel<<<1024, NTHR, 0, stream>>>(xbf, Wi, pi, l);
        scan_kernel<<<NBLK, NTHR, 0, stream>>>(pi, mask, bs, wsbf, wpbf, gates, cst,
                                               hst, hacc, rbuf, hbf, pbf, sync, l);
        finalize_kernel<<<1024, 256, 0, stream>>>(rbuf, fwd, bwd, out, l);
    }
    (void)in_sizes; (void)n_in; (void)out_size; (void)ws_size;
}
